// LaplacianModel_86062554678026
// MI455X (gfx1250) — matrix operand order repaired and checked by reading, not compiled, not run on hardware
//
#include <hip/hip_runtime.h>
#include <hip/hip_fp16.h>

typedef __attribute__((ext_vector_type(16))) _Float16 v16h;
typedef __attribute__((ext_vector_type(8)))  float    v8f;

#define HDIM 128
#define LSTRIDE (HDIM + 8)   // halves; pad to avoid LDS bank conflicts
#define ROWS 128             // MLP rows per block
#define NT 256               // threads per block (8 waves of 32)

// ---------------------------------------------------------------------------
// Branchless fast activations (CDNA5 has V_TANH_F32 as a TRANS32 op that can
// co-execute with WMMA; fall back to exp2/rcp which are also single TRANS ops)
// ---------------------------------------------------------------------------
__device__ __forceinline__ float fast_tanh(float x) {
#if __has_builtin(__builtin_amdgcn_tanhf)
  return __builtin_amdgcn_tanhf(x);
#else
  // tanh(x) = 1 - 2/(exp2(2*log2e*x)+1); clamp keeps exp2 in range, branchless
  float t = __builtin_amdgcn_exp2f(fminf(fmaxf(x, -16.f), 16.f) * 2.8853901f);
  return 1.0f - 2.0f * __builtin_amdgcn_rcpf(t + 1.0f);
#endif
}

__device__ __forceinline__ float fast_softplus(float z) {
  const float LOG2E = 1.4426950408889634f;
  const float LN2   = 0.6931471805599453f;
  float e = __builtin_amdgcn_exp2f(-fabsf(z) * LOG2E);          // v_exp_f32
  return fmaxf(z, 0.0f) + LN2 * __builtin_amdgcn_logf(1.0f + e); // v_log_f32
}

// ---------------------------------------------------------------------------
// Prep: W2 (f32, [K=128][N=128] row-major) -> w2t (f16, transposed [N][K])
// so that B-matrix fragments read contiguous K runs.
// ---------------------------------------------------------------------------
__global__ void k_w2_to_f16t(const float* __restrict__ W2,
                             _Float16* __restrict__ w2t) {
  int i = blockIdx.x * blockDim.x + threadIdx.x;
  if (i < HDIM * HDIM) {
    int k = i >> 7;
    int n = i & (HDIM - 1);
    w2t[n * HDIM + k] = (_Float16)W2[i];
  }
}

// ---------------------------------------------------------------------------
// Fused MLP: gather coordinates, layer1 (VALU, K=3), layer2 via WMMA f16,
// layer3 (N=1) + softplus. Row gr = b*17 + s; s: 0-7 original corners,
// 8-15 staggered corners, 16 center.
// ---------------------------------------------------------------------------
__global__ __launch_bounds__(NT)
void k_mlp(const float* __restrict__ DATAX, const float* __restrict__ STAGX,
           const int* __restrict__ CI,
           const float* __restrict__ W1, const float* __restrict__ b1,
           const _Float16* __restrict__ w2t, const float* __restrict__ b2,
           const float* __restrict__ W3, const float* __restrict__ b3v,
           float* __restrict__ OUT, int B, int nx) {
  __shared__ _Float16 sh1[ROWS][LSTRIDE];   // layer-1 activations, f16, [M][K]
  __shared__ _Float16 sw2[HDIM][LSTRIDE];   // W2^T, f16, [N][K]

  const int tid = threadIdx.x;
  const int ny = nx, nz = nx;
  const int N = nx * ny * nz;
  const int nys = ny - 1, nzs = nz - 1;
  const int Ns = (nx - 1) * nys * nzs;
  const long long rowBase = (long long)blockIdx.x * ROWS;
  const long long totalRows = 17LL * (long long)B;

  // Stage W2^T into LDS
  for (int i = tid; i < HDIM * HDIM; i += NT) {
    int n = i >> 7;
    int k = i & (HDIM - 1);
    sw2[n][k] = w2t[i];
  }

  // Layer 1: each thread computes half a row (64 of 128 hidden units)
  {
    int r  = tid >> 1;
    int c0 = (tid & 1) * 64;
    long long gr = rowBase + r;
    float x0 = 0.f, x1 = 0.f, x2 = 0.f;
    if (gr < totalRows) {
      int b = (int)(gr / 17);
      int s = (int)(gr % 17);
      int ci  = CI[b];
      int iz  = ci / (nx * ny);
      int rem = ci % (nx * ny);
      int iy  = rem / nx;
      int ix  = rem % nx;
      const float* src;
      if (s < 8) {                     // original-grid corner, d in {-1,1}, dz fastest
        int dx = (s & 4) ? 1 : -1;
        int dy = (s & 2) ? 1 : -1;
        int dz = (s & 1) ? 1 : -1;
        int fo = (iz + dz) * nx * ny + (iy + dy) * nx + (ix + dx);
        fo = min(max(fo, 0), N - 1);
        src = DATAX + 3LL * fo;
      } else if (s < 16) {             // staggered corner, d in {-1,0}
        int t  = s - 8;
        int dx = (t & 4) ? 0 : -1;
        int dy = (t & 2) ? 0 : -1;
        int dz = (t & 1) ? 0 : -1;
        int fs = (ix + dx) * nys * nzs + (iy + dy) * nzs + (iz + dz);
        fs = min(max(fs, 0), Ns - 1);
        src = STAGX + 3LL * fs;
      } else {                         // center point (prediction row)
        src = DATAX + 3LL * ci;
      }
      x0 = src[0]; x1 = src[1]; x2 = src[2];
    }
    #pragma unroll 8
    for (int j = 0; j < 64; ++j) {
      int c = c0 + j;
      float v = fmaf(x0, W1[0 * HDIM + c],
               fmaf(x1, W1[1 * HDIM + c],
               fmaf(x2, W1[2 * HDIM + c], b1[c])));
      sh1[r][c] = (_Float16)fast_tanh(v);
    }
  }
  __syncthreads();

  // Layer 2: WMMA. Each wave owns a 16-row strip, 8 column tiles, K=128 in 4 steps.
  const int wave = tid >> 5;
  const int lane = tid & 31;
  const int ncol = lane & 15;
  const int hi   = (lane >> 4) & 1;          // lane group 0: lanes 0-15, 1: lanes 16-31
  const int mrow = wave * 16 + ncol;         // A-fragment row for this lane

  // A fragments (ISA layout): elems 0-7 -> K = k0 + 8*hi + e,
  //                           elems 8-15 -> K = k0 + 16 + 8*hi + (e-8)
  v16h afrag[4];
  #pragma unroll
  for (int kk = 0; kk < 4; ++kk) {
    int kLo = kk * 32 + hi * 8;
    int kHi = kk * 32 + 16 + hi * 8;
    v16h a;
    #pragma unroll
    for (int e = 0; e < 8; ++e) a[e] = sh1[mrow][kLo + e];
    #pragma unroll
    for (int e = 0; e < 8; ++e) a[8 + e] = sh1[mrow][kHi + e];
    afrag[kk] = a;
  }

  float partial[8] = {0.f, 0.f, 0.f, 0.f, 0.f, 0.f, 0.f, 0.f};

  #pragma unroll
  for (int tn = 0; tn < 8; ++tn) {
    int n = tn * 16 + ncol;                  // B/D column for this lane
    v8f acc = {};
    #pragma unroll
    for (int kk = 0; kk < 4; ++kk) {
      // B fragment (ISA layout): elems 0-7 -> K = kk*32 + 8*hi + e, elems 8-15 -> K = kk*32 + 16 + 8*hi + (e-8), column n
      int kb = kk * 32 + hi * 8;
      v16h bfr;
      #pragma unroll
      for (int e = 0; e < 16; ++e) bfr[e] = sw2[n][kb + (e & 7) + ((e >> 3) << 4)];
      acc = __builtin_amdgcn_wmma_f32_16x16x32_f16(
          false, afrag[kk], false, bfr, (short)0, acc, false, false);
    }
    // Layer 3 partial: tanh(h2) * W3[n], accumulated per D-row r
    float bb2 = b2[n];
    float w3n = W3[n];
    #pragma unroll
    for (int r = 0; r < 8; ++r) {
      float h2 = fast_tanh(acc[r] + bb2);
      partial[r] = fmaf(h2, w3n, partial[r]);
    }
  }

  // Reduce the 16 lanes that share a row (xor <= 8 stays within each half-wave),
  // then softplus and store. D layout: VGPR r -> M = wave*16 + r + 8*hi.
  const float b3s = b3v[0];
  #pragma unroll
  for (int r = 0; r < 8; ++r) {
    float v = partial[r];
    v += __shfl_xor(v, 8, 32);
    v += __shfl_xor(v, 4, 32);
    v += __shfl_xor(v, 2, 32);
    v += __shfl_xor(v, 1, 32);
    if (ncol == 0) {
      long long gr = rowBase + wave * 16 + r + hi * 8;
      if (gr < totalRows) {
        OUT[gr] = fast_softplus(v + b3s);
      }
    }
  }
}

// ---------------------------------------------------------------------------
// Loss: per-center combine of the 17 MLP outputs -> per-block partial sums
// (sum dlap^2*m, sum m, sum |pred - dataf|). Deterministic tree reduction.
// ---------------------------------------------------------------------------
__global__ __launch_bounds__(256)
void k_loss(const float* __restrict__ OUT, const float* __restrict__ DATAF,
            const float* __restrict__ LAPLF, const int* __restrict__ CI,
            float* __restrict__ partials, int B, int nx, int nBlocks) {
  __shared__ float red[3][256];
  const int tid = threadIdx.x;
  const int b = blockIdx.x * 256 + tid;
  const int ny = nx, nz = nx;
  const int nym2 = ny - 2, nzm2 = nz - 2, nxm2 = nx - 2;
  float ssq = 0.f, sm = 0.f, sabs = 0.f;
  if (b < B) {
    int ci  = CI[b];
    int iz  = ci / (nx * ny);
    int rem = ci % (nx * ny);
    int iy  = rem / nx;
    int ix  = rem % nx;
    bool valid = (ix >= 1 && ix < nx - 1 && iy >= 1 && iy < ny - 1 &&
                  iz >= 1 && iz < nz - 1);
    const float* o = OUT + 17LL * b;
    float fo[8], fs[8];
    #pragma unroll
    for (int s = 0; s < 8; ++s) { fo[s] = o[s]; fs[s] = o[8 + s]; }
    const int ia[4]  = {0, 2, 1, 3};
    const int ibx[4] = {7, 5, 6, 4};
    float lap = 0.f;
    #pragma unroll
    for (int q = 0; q < 4; ++q) {
      float a  = fo[ia[q]] + fo[ibx[q]];
      float s2 = fs[ia[q]] + fs[ibx[q]];
      lap += fabsf(a - s2) / (a + s2);
    }
    lap *= (1.0f / 2.25f);   // / (3 * delta), delta = 3 * 0.5^2
    int tot = nxm2 * nym2 * nzm2;
    int fl  = (ix - 1) * nym2 * nzm2 + (iy - 1) * nzm2 + (iz - 1);
    fl = min(max(fl, 0), tot - 1);
    float dlap = LAPLF[fl] - lap;
    float m = valid ? 1.f : 0.f;
    ssq  = dlap * dlap * m;
    sm   = m;
    sabs = fabsf(o[16] - DATAF[ci]);
  }
  red[0][tid] = ssq; red[1][tid] = sm; red[2][tid] = sabs;
  __syncthreads();
  for (int s = 128; s > 0; s >>= 1) {
    if (tid < s) {
      red[0][tid] += red[0][tid + s];
      red[1][tid] += red[1][tid + s];
      red[2][tid] += red[2][tid + s];
    }
    __syncthreads();
  }
  if (tid == 0) {
    partials[blockIdx.x]               = red[0][0];
    partials[nBlocks + blockIdx.x]     = red[1][0];
    partials[2 * nBlocks + blockIdx.x] = red[2][0];
  }
}

__global__ __launch_bounds__(256)
void k_final(const float* __restrict__ partials, int nBlocks,
             const float* __restrict__ wgt, float* __restrict__ out, int B) {
  __shared__ float red[3][256];
  const int tid = threadIdx.x;
  float a = 0.f, b = 0.f, c = 0.f;
  for (int i = tid; i < nBlocks; i += 256) {
    a += partials[i];
    b += partials[nBlocks + i];
    c += partials[2 * nBlocks + i];
  }
  red[0][tid] = a; red[1][tid] = b; red[2][tid] = c;
  __syncthreads();
  for (int s = 128; s > 0; s >>= 1) {
    if (tid < s) {
      red[0][tid] += red[0][tid + s];
      red[1][tid] += red[1][tid + s];
      red[2][tid] += red[2][tid + s];
    }
    __syncthreads();
  }
  if (tid == 0) {
    float loss_m = wgt[0] * (red[0][0] / fmaxf(red[1][0], 1.0f));
    float loss_e = red[2][0] / (float)B;
    out[0] = loss_e + loss_m;
  }
}

// ---------------------------------------------------------------------------
extern "C" void kernel_launch(void* const* d_in, const int* in_sizes, int n_in,
                              void* d_out, int out_size, void* d_ws, size_t ws_size,
                              hipStream_t stream) {
  const float* DATAX = (const float*)d_in[0];
  const float* DATAF = (const float*)d_in[1];
  const float* STAGX = (const float*)d_in[2];
  const float* LAPLF = (const float*)d_in[3];
  const float* WGT   = (const float*)d_in[4];
  const float* W1    = (const float*)d_in[5];
  const float* b1    = (const float*)d_in[6];
  const float* W2    = (const float*)d_in[7];
  const float* b2    = (const float*)d_in[8];
  const float* W3    = (const float*)d_in[9];
  const float* b3    = (const float*)d_in[10];
  const int*   CI    = (const int*)d_in[11];

  long long N = (long long)in_sizes[0] / 3;
  int nx = 1;
  while ((long long)nx * nx * nx < N) ++nx;   // N = nx^3 (128 for reference)
  int B = in_sizes[11];

  char* ws = (char*)d_ws;
  _Float16* w2t = (_Float16*)ws;                                  // 32 KB
  float* OUT = (float*)(ws + (size_t)HDIM * HDIM * sizeof(_Float16));
  long long rows = 17LL * (long long)B;
  float* partials = (float*)(ws + (size_t)HDIM * HDIM * sizeof(_Float16)
                                + (size_t)rows * sizeof(float));
  int nBlocks = (B + 255) / 256;

  hipLaunchKernelGGL(k_w2_to_f16t, dim3((HDIM * HDIM + 255) / 256), dim3(256),
                     0, stream, W2, w2t);
  int mlpBlocks = (int)((rows + ROWS - 1) / ROWS);
  hipLaunchKernelGGL(k_mlp, dim3(mlpBlocks), dim3(NT), 0, stream,
                     DATAX, STAGX, CI, W1, b1, w2t, b2, W3, b3, OUT, B, nx);
  hipLaunchKernelGGL(k_loss, dim3(nBlocks), dim3(256), 0, stream,
                     OUT, DATAF, LAPLF, CI, partials, B, nx, nBlocks);
  hipLaunchKernelGGL(k_final, dim3(1), dim3(256), 0, stream,
                     partials, nBlocks, WGT, (float*)d_out, B);
}
